// FullAttention_87823491269160
// MI455X (gfx1250) — matrix operand order repaired and checked by reading, not compiled, not run on hardware
//
#include <hip/hip_runtime.h>
#include <hip/hip_bf16.h>

// ---------------------------------------------------------------------------
// Types
// ---------------------------------------------------------------------------
typedef __bf16 bf16_t;
typedef __attribute__((ext_vector_type(16))) __bf16 v16bf;
typedef __attribute__((ext_vector_type(8)))  float  v8f;
typedef __attribute__((ext_vector_type(4)))  unsigned int v4u;
typedef __attribute__((ext_vector_type(8)))  int v8i;
typedef __attribute__((ext_vector_type(4)))  int v4i;

// Problem constants (match reference)
constexpr int kHidden = 2048;
constexpr int kT      = 2048;
constexpr int kNQ     = 16;
constexpr int kNKV    = 4;
constexpr int kHD     = 128;
constexpr int kQGN    = 4096;   // 2 * NQ * HD
constexpr int kKVN    = 512;    // NKV * HD

// ---------------------------------------------------------------------------
// Tensor Data Mover: 2D tile (bf16) global -> LDS.
// D# group0: [1:0]=count=1, [63:32]=lds_addr, [120:64]=global_addr,
//            [127:126]=type=2.
// D# group1: [17:16]=data_size(1 => 2B), [79:48]=tensor_dim0,
//            [111:80]=tensor_dim1, [127:112]=tile_dim0, [143:128]=tile_dim1,
//            [207:160]=tensor_dim0_stride.
// Issue from one wave; completion via s_wait_tensorcnt.
// ---------------------------------------------------------------------------
__device__ inline void tdm_load_2d_bf16(unsigned int lds_addr, const void* gptr,
                                        unsigned int tile_w, unsigned int tile_h,
                                        unsigned int stride_elems) {
  unsigned long long ga = (unsigned long long)(uintptr_t)gptr;
  v4u g0;
  g0[0] = 1u;                                   // count=1 (valid user D#)
  g0[1] = lds_addr;                             // LDS byte address
  g0[2] = (unsigned int)ga;                     // global addr [95:64]
  g0[3] = (unsigned int)((ga >> 32) & 0x1FFFFFFu) | (2u << 30);  // addr hi | type=2
  v8i g1;
  g1[0] = (int)(1u << 16);                      // data_size = 1 -> 2 bytes
  g1[1] = (int)(tile_w << 16);                  // tensor_dim0 low16 (bits 63:48)
  g1[2] = (int)(tile_h << 16);                  // tensor_dim0 hi=0 | tensor_dim1 lo16
  g1[3] = (int)(tile_w << 16);                  // tensor_dim1 hi=0 | tile_dim0
  g1[4] = (int)tile_h;                          // tile_dim1 | tile_dim2=0
  g1[5] = (int)stride_elems;                    // tensor_dim0_stride low32
  g1[6] = 0;
  g1[7] = 0;
  v4i z4 = {0, 0, 0, 0};
#if __has_include(<hip/amd_detail/amd_gfx1250_TDM.h>)
  v8i z8 = {0, 0, 0, 0, 0, 0, 0, 0};
  __builtin_amdgcn_tensor_load_to_lds(g0, g1, z4, z4, z8, 0);
#else
  __builtin_amdgcn_tensor_load_to_lds(g0, g1, z4, z4, 0);
#endif
}

// ---------------------------------------------------------------------------
// WMMA fragment loaders (bf16, 16x16x32)
// A (16 rows x 32 K), row-major tile with given stride (elems):
//   lane<16:  elems 0..7 -> K0..7,  elems 8..15 -> K16..23
//   lane>=16: elems 0..7 -> K8..15, elems 8..15 -> K24..31
// B (32 K x 16 cols) fed from TRANSPOSED tile [n][k]:
//   lane<16:  elems 0..7 -> K0..7,  elems 8..15 -> K16..23
//   lane>=16: elems 0..7 -> K8..15, elems 8..15 -> K24..31; col = lane&15
// ---------------------------------------------------------------------------
__device__ inline v16bf load_fragA(const bf16_t* tile, int stride, int lane) {
  int row = lane & 15;
  int kh  = (lane & 16) ? 8 : 0;
  const bf16_t* p = tile + (size_t)row * stride + kh;
  v16bf f;
  ((uint4*)&f)[0] = *(const uint4*)(p);
  ((uint4*)&f)[1] = *(const uint4*)(p + 16);
  return f;
}

__device__ inline v16bf load_fragB(const bf16_t* tileT, int stride, int lane) {
  int col = lane & 15;
  int kh  = (lane & 16) ? 8 : 0;
  const bf16_t* p = tileT + (size_t)col * stride + kh;
  v16bf f;
  ((uint4*)&f)[0] = *(const uint4*)(p);
  ((uint4*)&f)[1] = *(const uint4*)(p + 16);
  return f;
}

__device__ inline float half_reduce_max(float x) {
#pragma unroll
  for (int m = 1; m <= 8; m <<= 1) x = fmaxf(x, __shfl_xor(x, m, 32));
  return x;
}
__device__ inline float half_reduce_add(float x) {
#pragma unroll
  for (int m = 1; m <= 8; m <<= 1) x += __shfl_xor(x, m, 32);
  return x;
}

// ---------------------------------------------------------------------------
// f32 -> bf16 elementwise
// ---------------------------------------------------------------------------
__global__ __launch_bounds__(256)
void cvt_bf16_kernel(const float* __restrict__ src, bf16_t* __restrict__ dst, size_t n) {
  size_t i = (size_t)blockIdx.x * blockDim.x + threadIdx.x;
  if (i < n) dst[i] = (bf16_t)src[i];
}

// ---------------------------------------------------------------------------
// Tiled bf16 GEMM: C[M,N] = A[M,K] * B[K,N], A/B row-major bf16, C f32.
// Block: 256 threads (8 waves). Block tile 128x128, K-step 32.
// Wave tile 32x64: 2x4 grid of 16x16 WMMA -> 8 v_wmma per K-step per wave.
// A tile (strided 2D) fetched by the Tensor Data Mover from wave 0;
// B tile transposed into LDS by all threads.
// ---------------------------------------------------------------------------
__global__ __launch_bounds__(256)
void gemm_bf16_kernel(const bf16_t* __restrict__ A, const bf16_t* __restrict__ B,
                      float* __restrict__ C, int M, int N, int K) {
  __shared__ bf16_t lA[128][32];    // 8 KB, filled by TDM
  __shared__ bf16_t lBt[128][32];   // 8 KB (B tile transposed: [n][k])

  const int bm   = blockIdx.y * 128;
  const int bn   = blockIdx.x * 128;
  const int tid  = threadIdx.x;
  const int lane = tid & 31;
  const int wave = tid >> 5;
  const int wm   = (wave & 3) * 32;
  const int wn   = (wave >> 2) * 64;

  const unsigned int ldsA = (unsigned int)(uintptr_t)&lA[0][0];

  v8f acc[2][4] = {};

  for (int k0 = 0; k0 < K; k0 += 32) {
    // A tile 128x32 (row stride K elems) via TDM, issued by wave 0
    if (tid < 32) {
      tdm_load_2d_bf16(ldsA, A + (size_t)bm * K + k0, 32, 128, (unsigned)K);
      __builtin_amdgcn_s_wait_tensorcnt(0);
    }
    // B tile 32x128, write transposed into lBt[n][k]
    {
      int kk = tid >> 3;             // 0..31
      int n0 = (tid & 7) * 16;       // 0..112
      const bf16_t* src = B + (size_t)(k0 + kk) * N + bn + n0;
      uint4 t0 = *(const uint4*)(src);
      uint4 t1 = *(const uint4*)(src + 8);
      const bf16_t* e0 = (const bf16_t*)&t0;
      const bf16_t* e1 = (const bf16_t*)&t1;
#pragma unroll
      for (int i = 0; i < 8; ++i) lBt[n0 + i][kk] = e0[i];
#pragma unroll
      for (int i = 0; i < 8; ++i) lBt[n0 + 8 + i][kk] = e1[i];
    }
    __syncthreads();

#pragma unroll
    for (int i = 0; i < 2; ++i) {
      v16bf a = load_fragA(&lA[wm + i * 16][0], 32, lane);
#pragma unroll
      for (int j = 0; j < 4; ++j) {
        v16bf b = load_fragB(&lBt[wn + j * 16][0], 32, lane);
        acc[i][j] = __builtin_amdgcn_wmma_f32_16x16x32_bf16(
            false, a, false, b, (short)0, acc[i][j], false, false);
      }
    }
    __syncthreads();
  }

  const int col15 = lane & 15;
  const int half8 = (lane & 16) ? 8 : 0;
#pragma unroll
  for (int i = 0; i < 2; ++i)
#pragma unroll
    for (int j = 0; j < 4; ++j) {
      int col = bn + wn + j * 16 + col15;
      int r0  = bm + wm + i * 16 + half8;
#pragma unroll
      for (int v = 0; v < 8; ++v)
        C[(size_t)(r0 + v) * N + col] = acc[i][j][v];
    }
}

// ---------------------------------------------------------------------------
// RMSNorm + RoPE over heads, relayout to head-major bf16.
// blockIdx.x = t, blockIdx.y = head slot: 0-15 Q, 16-19 K, 20-23 V(copy only)
// 128 threads = one per d.
// ---------------------------------------------------------------------------
__global__ __launch_bounds__(128)
void norm_rope_kernel(const float* __restrict__ qg, const float* __restrict__ kf,
                      const float* __restrict__ vf, const int* __restrict__ pos_ids,
                      const float* __restrict__ inv_freq,
                      const float* __restrict__ qw, const float* __restrict__ kw,
                      bf16_t* __restrict__ q_bf, bf16_t* __restrict__ k_bf,
                      bf16_t* __restrict__ v_bf) {
  const int t = blockIdx.x;
  const int h = blockIdx.y;
  const int d = threadIdx.x;
  __shared__ float xs[128];
  __shared__ float red[4];

  float val;
  if (h < 16) {
    val = qg[(size_t)t * kQGN + h * kHD + d];
  } else if (h < 20) {
    val = kf[(size_t)t * kKVN + (h - 16) * kHD + d];
  } else {
    val = vf[(size_t)t * kKVN + (h - 20) * kHD + d];
    v_bf[((size_t)(h - 20) * kT + t) * kHD + d] = (bf16_t)val;
    return;
  }
  xs[d] = val;

  float ss = val * val;
#pragma unroll
  for (int m = 1; m <= 16; m <<= 1) ss += __shfl_xor(ss, m, 32);
  if ((d & 31) == 0) red[d >> 5] = ss;
  __syncthreads();
  ss = red[0] + red[1] + red[2] + red[3];
  float s = rsqrtf(ss * (1.0f / 128.0f) + 1e-6f);

  const float* w = (h < 16) ? qw : kw;
  float xn  = val * s * w[d];
  int   dp  = (d < 64) ? d + 64 : d - 64;
  float xpn = xs[dp] * s * w[dp];

  float pos = (float)pos_ids[t];
  float fr  = pos * inv_freq[d & 63];
  float c = __cosf(fr), sn = __sinf(fr);
  float rot  = (d < 64) ? -xpn : xpn;
  float outv = xn * c + rot * sn;

  if (h < 16) q_bf[((size_t)h * kT + t) * kHD + d]        = (bf16_t)outv;
  else        k_bf[((size_t)(h - 16) * kT + t) * kHD + d] = (bf16_t)outv;
}

// ---------------------------------------------------------------------------
// Flash attention, causal, GQA. Grid (T/64, NQ), block 128 (4 waves).
// Wave w: 16 query rows starting at qb + 16*w.
// K tile (32x128, contiguous) fetched by TDM; V tile transposed manually.
// QK^T and P*V via v_wmma_f32_16x16x32_bf16 with online softmax.
// ---------------------------------------------------------------------------
__global__ __launch_bounds__(128)
void attn_kernel(const bf16_t* __restrict__ q_bf, const bf16_t* __restrict__ k_bf,
                 const bf16_t* __restrict__ v_bf, float* __restrict__ attnO) {
  __shared__ bf16_t lK[32][128];   // [key][d]           8 KB (TDM target)
  __shared__ bf16_t lVt[128][32];  // [d][key]           8 KB
  __shared__ bf16_t lP[4][16][32]; // per-wave P staging  4 KB

  const int h    = blockIdx.y;
  const int kvh  = h >> 2;
  const int qb   = blockIdx.x * 64;
  const int tid  = threadIdx.x;
  const int lane = tid & 31;
  const int w    = tid >> 5;
  const int qrow0 = qb + w * 16;
  const int col15 = lane & 15;
  const int half8 = (lane & 16) ? 8 : 0;
  const float scale = 0.08838834764831845f;  // 1/sqrt(128)

  const unsigned int ldsK = (unsigned int)(uintptr_t)&lK[0][0];

  // Preload Q fragments (4 chunks of K=32 along d)
  v16bf aq[4];
  const bf16_t* qbase = q_bf + ((size_t)h * kT + qrow0) * kHD;
#pragma unroll
  for (int c = 0; c < 4; ++c) aq[c] = load_fragA(qbase + c * 32, kHD, lane);

  v8f o[8] = {};
  float m[8], l[8], alphav[8];
#pragma unroll
  for (int v = 0; v < 8; ++v) { m[v] = -3.0e38f; l[v] = 0.0f; }

  const int ntiles = qb / 32 + 2;  // keys 0 .. qb+63 inclusive
  for (int kt = 0; kt < ntiles; ++kt) {
    // K tile (rows kt*32..+32, full 128-wide rows, contiguous) via TDM
    if (tid < 32) {
      tdm_load_2d_bf16(ldsK, k_bf + ((size_t)kvh * kT + kt * 32) * kHD,
                       128, 32, 128);
      __builtin_amdgcn_s_wait_tensorcnt(0);
    }
    // V tile transposed into lVt[d][key]
    {
      int r  = tid >> 2;
      int c0 = (tid & 3) * 32;
      const bf16_t* src = v_bf + ((size_t)kvh * kT + kt * 32 + r) * kHD + c0;
#pragma unroll
      for (int i = 0; i < 32; ++i) lVt[c0 + i][r] = src[i];
    }
    __syncthreads();

    // S = Q * K^T for two 16-key sub-tiles (8 WMMA)
    v8f s2[2] = {};
#pragma unroll
    for (int c = 0; c < 4; ++c) {
#pragma unroll
      for (int j = 0; j < 2; ++j) {
        v16bf b = load_fragB(&lK[j * 16][c * 32], kHD, lane);
        s2[j] = __builtin_amdgcn_wmma_f32_16x16x32_bf16(
            false, aq[c], false, b, (short)0, s2[j], false, false);
      }
    }

    // scale + causal mask
#pragma unroll
    for (int j = 0; j < 2; ++j) {
      int key = kt * 32 + j * 16 + col15;
#pragma unroll
      for (int v = 0; v < 8; ++v) {
        float sv = s2[j][v] * scale;
        int row  = qrow0 + half8 + v;
        s2[j][v] = (key <= row) ? sv : -3.0e38f;
      }
    }

    // online softmax per row (row = half8 + v); reduce across 16-lane half
#pragma unroll
    for (int v = 0; v < 8; ++v) {
      float cand = fmaxf(s2[0][v], s2[1][v]);
      cand = half_reduce_max(cand);
      float mn    = fmaxf(m[v], cand);
      float alpha = __expf(m[v] - mn);
      float p0 = __expf(s2[0][v] - mn);
      float p1 = __expf(s2[1][v] - mn);
      float rs = half_reduce_add(p0 + p1);
      l[v] = l[v] * alpha + rs;
      m[v] = mn;
      alphav[v] = alpha;
      lP[w][half8 + v][col15]      = (bf16_t)p0;
      lP[w][half8 + v][16 + col15] = (bf16_t)p1;
    }

    // rescale accumulators
#pragma unroll
    for (int n = 0; n < 8; ++n)
#pragma unroll
      for (int v = 0; v < 8; ++v) o[n][v] *= alphav[v];

    // O += P * V  (8 WMMA over d-chunks of 16)
    v16bf pa = load_fragA(&lP[w][0][0], 32, lane);
#pragma unroll
    for (int n = 0; n < 8; ++n) {
      v16bf b = load_fragB(&lVt[n * 16][0], 32, lane);
      o[n] = __builtin_amdgcn_wmma_f32_16x16x32_bf16(
          false, pa, false, b, (short)0, o[n], false, false);
    }
    __syncthreads();
  }

  // epilogue: normalize and store [t][NQ*HD]
#pragma unroll
  for (int n = 0; n < 8; ++n) {
#pragma unroll
    for (int v = 0; v < 8; ++v) {
      int row = qrow0 + half8 + v;
      attnO[(size_t)row * (kNQ * kHD) + h * kHD + n * 16 + col15] = o[n][v] / l[v];
    }
  }
}

// ---------------------------------------------------------------------------
// SiLU gate: gated_bf16 = attnO * silu(gate)  (gate = qg[:, 2048:4096])
// ---------------------------------------------------------------------------
__global__ __launch_bounds__(256)
void gate_kernel(const float* __restrict__ attnO, const float* __restrict__ qg,
                 bf16_t* __restrict__ gated) {
  size_t i = (size_t)blockIdx.x * blockDim.x + threadIdx.x;  // over T * 2048
  int t = (int)(i >> 11);
  int c = (int)(i & 2047);
  float g  = qg[(size_t)t * kQGN + 2048 + c];
  float sg = g / (1.0f + __expf(-g));
  gated[i] = (bf16_t)(attnO[i] * sg);
}

// ---------------------------------------------------------------------------
// Host launch
// ---------------------------------------------------------------------------
extern "C" void kernel_launch(void* const* d_in, const int* in_sizes, int n_in,
                              void* d_out, int out_size, void* d_ws, size_t ws_size,
                              hipStream_t stream) {
  const float* x        = (const float*)d_in[0];
  const int*   pos_ids  = (const int*)d_in[1];
  const float* inv_freq = (const float*)d_in[2];
  const float* Wq       = (const float*)d_in[3];
  const float* Wk       = (const float*)d_in[4];
  const float* Wv       = (const float*)d_in[5];
  const float* Wo       = (const float*)d_in[6];
  const float* qw       = (const float*)d_in[7];
  const float* kw       = (const float*)d_in[8];
  float* out = (float*)d_out;

  char*  ws  = (char*)d_ws;
  size_t cur = 0;
  auto alloc = [&](size_t bytes) -> char* {
    char* p = ws + cur;
    cur += (bytes + 255) & ~(size_t)255;
    return p;
  };

  const size_t nx  = (size_t)kT * kHidden;
  const size_t nwq = (size_t)kHidden * kQGN;
  const size_t nwk = (size_t)kHidden * kKVN;
  const size_t nwo = (size_t)(kNQ * kHD) * kHidden;

  bf16_t* x_bf  = (bf16_t*)alloc(nx * 2);
  bf16_t* Wq_bf = (bf16_t*)alloc(nwq * 2);
  bf16_t* Wk_bf = (bf16_t*)alloc(nwk * 2);
  bf16_t* Wv_bf = (bf16_t*)alloc(nwk * 2);
  bf16_t* Wo_bf = (bf16_t*)alloc(nwo * 2);
  float*  qg_f  = (float*)alloc((size_t)kT * kQGN * 4);
  float*  k_f   = (float*)alloc((size_t)kT * kKVN * 4);
  float*  v_f   = (float*)alloc((size_t)kT * kKVN * 4);
  bf16_t* q_bf  = (bf16_t*)alloc((size_t)kNQ * kT * kHD * 2);
  bf16_t* k_bfh = (bf16_t*)alloc((size_t)kNKV * kT * kHD * 2);
  bf16_t* v_bfh = (bf16_t*)alloc((size_t)kNKV * kT * kHD * 2);
  float*  attnf = (float*)alloc((size_t)kT * (kNQ * kHD) * 4);
  bf16_t* gated = (bf16_t*)alloc((size_t)kT * (kNQ * kHD) * 2);

  auto cvt = [&](const float* s, bf16_t* d, size_t n) {
    cvt_bf16_kernel<<<(unsigned)((n + 255) / 256), 256, 0, stream>>>(s, d, n);
  };
  cvt(x, x_bf, nx);
  cvt(Wq, Wq_bf, nwq);
  cvt(Wk, Wk_bf, nwk);
  cvt(Wv, Wv_bf, nwk);
  cvt(Wo, Wo_bf, nwo);

  // Projections (block tile 128x128)
  gemm_bf16_kernel<<<dim3(kQGN / 128, kT / 128), 256, 0, stream>>>(
      x_bf, Wq_bf, qg_f, kT, kQGN, kHidden);
  gemm_bf16_kernel<<<dim3(kKVN / 128, kT / 128), 256, 0, stream>>>(
      x_bf, Wk_bf, k_f, kT, kKVN, kHidden);
  gemm_bf16_kernel<<<dim3(kKVN / 128, kT / 128), 256, 0, stream>>>(
      x_bf, Wv_bf, v_f, kT, kKVN, kHidden);

  // RMSNorm + RoPE + relayout (Q:0-15, K:16-19, V copy:20-23)
  norm_rope_kernel<<<dim3(kT, 24), 128, 0, stream>>>(
      qg_f, k_f, v_f, pos_ids, inv_freq, qw, kw, q_bf, k_bfh, v_bfh);

  // Flash attention
  attn_kernel<<<dim3(kT / 64, kNQ), 128, 0, stream>>>(q_bf, k_bfh, v_bfh, attnf);

  // SiLU gate
  gate_kernel<<<(unsigned)(((size_t)kT * 2048) / 256), 256, 0, stream>>>(
      attnf, qg_f, gated);

  // Output projection
  gemm_bf16_kernel<<<dim3(kHidden / 128, kT / 128), 256, 0, stream>>>(
      gated, Wo_bf, out, kT, kHidden, kNQ * kHD);
}
